// Smaill_26182120636648
// MI455X (gfx1250) — matrix operand order repaired and checked by reading, not compiled, not run on hardware
//
#include <hip/hip_runtime.h>
#include <hip/hip_bf16.h>
#include <math.h>

// ---------------------------------------------------------------------------
// GPT forward for MI455X (gfx1250, wave32). All GEMMs via v_wmma_f32_16x16x32_bf16
// with on-the-fly f32->bf16 conversion in LDS staging; f32 accumulate/epilogue.
// Block tile 128x64, BK=32, 4 waves (each 32 rows x 64 cols = 2x4 WMMA frags),
// double-buffered LDS, one barrier per K-step. All shapes are multiples of the
// tile sizes (asserted by construction in kernel_launch), so no bounds checks.
// ---------------------------------------------------------------------------

typedef __attribute__((ext_vector_type(16))) __bf16 bf16x16;
typedef __attribute__((ext_vector_type(8)))  __bf16 bf16x8;
typedef __attribute__((ext_vector_type(4)))  __bf16 bf16x4;
typedef __attribute__((ext_vector_type(8)))  float  f32x8;

#define GF_ADD    1   // C += result (residual)
#define GF_BIAS   2   // add bias[col]
#define GF_GELU   4   // exact gelu epilogue
#define GF_TRANSB 8   // B is stored [N x K] (A @ B^T)

#define BM 128
#define BN 64
#define BK 32
// LDS row pitch (elements): keeps 16B alignment for b128 fragment loads and
// breaks bank conflicts (40*2B = 80B, multiple of 16).
#define LPITCH 40

// ---- staging helpers: global f32 -> regs -> bf16 LDS ----------------------
__device__ __forceinline__
void stage_load_A(const float* __restrict__ A, float4* reg,
                  int rowBase, int k0, int lda, int tid)
{
    #pragma unroll
    for (int i = 0; i < 8; i++) {                 // 128x32 /4 /128thr = 8
        int c = i * 128 + tid;
        int r = c >> 3, kk = (c & 7) * 4;
        reg[i] = *(const float4*)(A + (size_t)(rowBase + r) * lda + k0 + kk);
    }
}
__device__ __forceinline__
void stage_store_A(const float4* reg, __bf16* dst, int tid)
{
    #pragma unroll
    for (int i = 0; i < 8; i++) {
        int c = i * 128 + tid;
        int r = c >> 3, kk = (c & 7) * 4;
        bf16x4 p = { (__bf16)reg[i].x, (__bf16)reg[i].y,
                     (__bf16)reg[i].z, (__bf16)reg[i].w };
        *(bf16x4*)(dst + r * LPITCH + kk) = p;    // ds_store_b64
    }
}
__device__ __forceinline__
void stage_load_B(const float* __restrict__ B, float4* reg,
                  int colBase, int k0, int ldb, int tid, bool transB)
{
    if (transB) {                                  // B[N][K], k contiguous
        #pragma unroll
        for (int i = 0; i < 4; i++) {              // 64x32 /4 /128 = 4
            int c = i * 128 + tid;
            int n = c >> 3, kk = (c & 7) * 4;
            reg[i] = *(const float4*)(B + (size_t)(colBase + n) * ldb + k0 + kk);
        }
    } else {                                       // B[K][N], n contiguous
        #pragma unroll
        for (int i = 0; i < 4; i++) {
            int c = i * 128 + tid;
            int n4 = c & 15, kk = c >> 4;
            reg[i] = *(const float4*)(B + (size_t)(k0 + kk) * ldb + colBase + n4 * 4);
        }
    }
}
__device__ __forceinline__
void stage_store_B(const float4* reg, __bf16* dst, int tid, bool transB)
{
    if (transB) {
        #pragma unroll
        for (int i = 0; i < 4; i++) {
            int c = i * 128 + tid;
            int n = c >> 3, kk = (c & 7) * 4;
            bf16x4 p = { (__bf16)reg[i].x, (__bf16)reg[i].y,
                         (__bf16)reg[i].z, (__bf16)reg[i].w };
            *(bf16x4*)(dst + n * LPITCH + kk) = p;
        }
    } else {                                       // transpose into Bs[n][k]
        #pragma unroll
        for (int i = 0; i < 4; i++) {
            int c = i * 128 + tid;
            int n4 = c & 15, kk = c >> 4;
            int n = n4 * 4;
            dst[(n + 0) * LPITCH + kk] = (__bf16)reg[i].x;
            dst[(n + 1) * LPITCH + kk] = (__bf16)reg[i].y;
            dst[(n + 2) * LPITCH + kk] = (__bf16)reg[i].z;
            dst[(n + 3) * LPITCH + kk] = (__bf16)reg[i].w;
        }
    }
}

__global__ __launch_bounds__(128)
void gemm_wmma(const float* __restrict__ A, const float* __restrict__ Bm,
               float* __restrict__ Cm, const float* __restrict__ bias,
               int M, int N, int K, int lda, int ldb, int ldc,
               long long aStrideZ, long long bStrideZ, long long cStrideZ,
               int flags)
{
    A  += (long long)blockIdx.z * aStrideZ;
    Bm += (long long)blockIdx.z * bStrideZ;
    Cm += (long long)blockIdx.z * cStrideZ;
    const bool transB = (flags & GF_TRANSB) != 0;

    __shared__ __bf16 As[2][BM * LPITCH];   // [row][k], k contiguous
    __shared__ __bf16 Bs[2][BN * LPITCH];   // [col][k] (B transposed into LDS)

    const int tid  = threadIdx.x;
    const int lane = tid & 31;
    const int wave = tid >> 5;              // 4 waves -> rows wave*32..+31
    const int lhi  = lane >> 4;             // 0/1 half-wave
    const int ll   = lane & 15;

    const int rowBase = blockIdx.y * BM;
    const int colBase = blockIdx.x * BN;

    f32x8 acc[2][4];
    #pragma unroll
    for (int i = 0; i < 2; i++)
        #pragma unroll
        for (int j = 0; j < 4; j++)
            #pragma unroll
            for (int r = 0; r < 8; r++) acc[i][j][r] = 0.0f;

    float4 aReg[8], bReg[4];

    // prologue: stage tile 0
    stage_load_A(A, aReg, rowBase, 0, lda, tid);
    stage_load_B(Bm, bReg, colBase, 0, ldb, tid, transB);
    stage_store_A(aReg, &As[0][0], tid);
    stage_store_B(bReg, &Bs[0][0], tid, transB);
    __syncthreads();

    int cur = 0;
    for (int k0 = 0; k0 < K; k0 += BK) {
        const bool hasNext = (k0 + BK) < K;
        if (hasNext) {                       // overlap global loads with WMMA
            stage_load_A(A, aReg, rowBase, k0 + BK, lda, tid);
            stage_load_B(Bm, bReg, colBase, k0 + BK, ldb, tid, transB);
        }

        // ---- fragments per ISA VGPR layouts ----
        // A (16x32): lane ll = M row; elem j: k = lhi*8+j (j<8), 16+lhi*8+(j-8)
        // B (32x16): lane ll = N col; elem j: k = lhi*8+j (j<8), 16+lhi*8+(j-8)
        union FA { bf16x16 v; bf16x8 h[2]; } fa[2];
        union FB { bf16x16 v; bf16x8 h[2]; } fb[4];
        #pragma unroll
        for (int mt = 0; mt < 2; mt++) {
            int r = wave * 32 + mt * 16 + ll;
            fa[mt].h[0] = *(const bf16x8*)&As[cur][r * LPITCH + lhi * 8];
            fa[mt].h[1] = *(const bf16x8*)&As[cur][r * LPITCH + 16 + lhi * 8];
        }
        #pragma unroll
        for (int nt = 0; nt < 4; nt++) {
            int c = nt * 16 + ll;
            fb[nt].h[0] = *(const bf16x8*)&Bs[cur][c * LPITCH + lhi * 8];
            fb[nt].h[1] = *(const bf16x8*)&Bs[cur][c * LPITCH + 16 + lhi * 8];
        }
        #pragma unroll
        for (int mt = 0; mt < 2; mt++)
            #pragma unroll
            for (int nt = 0; nt < 4; nt++)
                acc[mt][nt] = __builtin_amdgcn_wmma_f32_16x16x32_bf16(
                    false, fa[mt].v, false, fb[nt].v,
                    (short)0, acc[mt][nt], false, false);

        if (hasNext) {
            stage_store_A(aReg, &As[cur ^ 1][0], tid);
            stage_store_B(bReg, &Bs[cur ^ 1][0], tid, transB);
            __syncthreads();
            cur ^= 1;
        }
    }

    // ---- epilogue: C layout lane=(n), VGPR r -> m = lhi*8 + r ----
    #pragma unroll
    for (int nt = 0; nt < 4; nt++) {
        const int col = colBase + nt * 16 + ll;
        const float bv = (flags & GF_BIAS) ? bias[col] : 0.0f;
        #pragma unroll
        for (int mt = 0; mt < 2; mt++) {
            #pragma unroll
            for (int r = 0; r < 8; r++) {
                int row = rowBase + wave * 32 + mt * 16 + lhi * 8 + r;
                float v = acc[mt][nt][r] + bv;
                if (flags & GF_GELU)
                    v = 0.5f * v * (1.0f + erff(v * 0.70710678118654752f));
                size_t o = (size_t)row * ldc + col;
                if (flags & GF_ADD) Cm[o] += v; else Cm[o] = v;
            }
        }
    }
}

// ---------------- LayerNorm: one block (256 thr) per row, C = 1024 ----------
__global__ __launch_bounds__(256)
void layernorm_k(const float* __restrict__ x, const float* __restrict__ g,
                 const float* __restrict__ b, float* __restrict__ y, int C)
{
    int row = blockIdx.x;
    const float* xr = x + (size_t)row * C;
    float* yr       = y + (size_t)row * C;
    __shared__ float red[256];

    float local[4];
    float s = 0.0f;
    #pragma unroll
    for (int i = 0; i < 4; i++) {
        int c = threadIdx.x + 256 * i;
        local[i] = (c < C) ? xr[c] : 0.0f;
        s += local[i];
    }
    red[threadIdx.x] = s; __syncthreads();
    for (int st = 128; st > 0; st >>= 1) {
        if (threadIdx.x < st) red[threadIdx.x] += red[threadIdx.x + st];
        __syncthreads();
    }
    float mean = red[0] / (float)C; __syncthreads();

    float s2 = 0.0f;
    #pragma unroll
    for (int i = 0; i < 4; i++) {
        int c = threadIdx.x + 256 * i;
        float d = (c < C) ? (local[i] - mean) : 0.0f;
        s2 += d * d;
    }
    red[threadIdx.x] = s2; __syncthreads();
    for (int st = 128; st > 0; st >>= 1) {
        if (threadIdx.x < st) red[threadIdx.x] += red[threadIdx.x + st];
        __syncthreads();
    }
    float rstd = rsqrtf(red[0] / (float)C + 1e-5f);
    #pragma unroll
    for (int i = 0; i < 4; i++) {
        int c = threadIdx.x + 256 * i;
        if (c < C) yr[c] = (local[i] - mean) * rstd * g[c] + b[c];
    }
}

// -------- causal softmax, in place on [H][T][T] scores for one batch b ------
__global__ __launch_bounds__(256)
void softmax_causal_k(float* __restrict__ S, int T, float scale)
{
    int rowIdx = blockIdx.x;          // h*T + t
    int t = rowIdx % T;
    float* p = S + (size_t)rowIdx * T;
    __shared__ float red[256];

    float v[4];
    float mx = -3.0e38f;
    #pragma unroll
    for (int i = 0; i < 4; i++) {
        int u = threadIdx.x + 256 * i;
        float s = (u < T) ? p[u] : 0.0f;
        v[i] = (u <= t) ? s * scale : -3.0e38f;
        mx = fmaxf(mx, v[i]);
    }
    red[threadIdx.x] = mx; __syncthreads();
    for (int st = 128; st > 0; st >>= 1) {
        if (threadIdx.x < st)
            red[threadIdx.x] = fmaxf(red[threadIdx.x], red[threadIdx.x + st]);
        __syncthreads();
    }
    mx = red[0]; __syncthreads();

    float sum = 0.0f;
    #pragma unroll
    for (int i = 0; i < 4; i++) {
        int u = threadIdx.x + 256 * i;
        float e = (u <= t) ? expf(v[i] - mx) : 0.0f;
        v[i] = e; sum += e;
    }
    red[threadIdx.x] = sum; __syncthreads();
    for (int st = 128; st > 0; st >>= 1) {
        if (threadIdx.x < st) red[threadIdx.x] += red[threadIdx.x + st];
        __syncthreads();
    }
    float inv = 1.0f / red[0];
    #pragma unroll
    for (int i = 0; i < 4; i++) {
        int u = threadIdx.x + 256 * i;
        if (u < T) p[u] = v[i] * inv;   // zeros above the diagonal
    }
}

// ---------------- embedding gather: x = tok_emb[idx] + pos_emb --------------
__global__ __launch_bounds__(256)
void embed_k(const int* __restrict__ idx, const float* __restrict__ tok,
             const float* __restrict__ pos, float* __restrict__ x, int T, int C)
{
    int row = blockIdx.x;             // b*T + t
    int t = row % T;
    int token = idx[row];
    const float* te = tok + (size_t)token * C;
    const float* pe = pos + (size_t)t * C;
    float* xr = x + (size_t)row * C;
    for (int c = threadIdx.x; c < C; c += blockDim.x) xr[c] = te[c] + pe[c];
}

// ---------------- sum Ph_b over heads: out[c] = sum_h phb[h][c] -------------
__global__ __launch_bounds__(256)
void headbias_sum_k(const float* __restrict__ phb, float* __restrict__ out,
                    int H, int C)
{
    int c = blockIdx.x * blockDim.x + threadIdx.x;
    if (c < C) {
        float s = 0.0f;
        for (int h = 0; h < H; h++) s += phb[(size_t)h * C + c];
        out[c] = s;
    }
}

// ---------------------------------------------------------------------------
extern "C" void kernel_launch(void* const* d_in, const int* in_sizes, int n_in,
                              void* d_out, int out_size, void* d_ws, size_t ws_size,
                              hipStream_t stream)
{
    (void)in_sizes; (void)n_in; (void)out_size;
    const int V = 32000, Bb = 4, T = 1024, C = 1024, H = 16, L = 8, HS = 64;
    const int R = Bb * T;                      // 4096 rows of the stream
    const float scale = 0.03125f;              // C^-0.5

    const int*   idx    = (const int*)  d_in[0];
    const float* tok    = (const float*)d_in[1];
    const float* pos    = (const float*)d_in[2];
    const float* Wq     = (const float*)d_in[3];
    const float* Wk     = (const float*)d_in[4];
    const float* Wv     = (const float*)d_in[5];
    const float* Ph_w   = (const float*)d_in[6];
    const float* Ph_b   = (const float*)d_in[7];
    const float* Pm_w   = (const float*)d_in[8];
    const float* Pm_b   = (const float*)d_in[9];
    const float* ln1_g  = (const float*)d_in[10];
    const float* ln1_b  = (const float*)d_in[11];
    const float* ln2_g  = (const float*)d_in[12];
    const float* ln2_b  = (const float*)d_in[13];
    const float* W1     = (const float*)d_in[14];
    const float* b1     = (const float*)d_in[15];
    const float* W2     = (const float*)d_in[16];
    const float* b2     = (const float*)d_in[17];
    const float* lnf_g  = (const float*)d_in[18];
    const float* lnf_b  = (const float*)d_in[19];
    const float* head_w = (const float*)d_in[20];
    const float* head_b = (const float*)d_in[21];
    float* logits = (float*)d_out;

    // ---- workspace partition (all f32) ----
    const size_t SZ_STREAM = (size_t)R * C;          // 4096*1024
    const size_t SZ_QKV    = (size_t)R * C;          // [H][R][HS] = same count
    const size_t SZ_H1     = (size_t)R * 4 * C;      // 4096*4096
    const size_t SZ_SC     = (size_t)H * T * T;      // per-batch scores
    float* ws = (float*)d_ws;
    size_t off = 0;
    float* x      = ws + off; off += SZ_STREAM;
    float* xn     = ws + off; off += SZ_STREAM;
    float* q      = ws + off; off += SZ_QKV;
    float* k      = ws + off; off += SZ_QKV;
    float* v      = ws + off; off += SZ_QKV;
    float* o_cat  = ws + off; off += SZ_STREAM;
    float* sa_pre = ws + off; off += SZ_STREAM;
    float* h1     = ws + off; off += SZ_H1;
    float* scores = ws + off; off += SZ_SC;
    float* bsum   = ws + off; off += 1024;
    if (off * sizeof(float) > ws_size) return;       // workspace guard

    // ---- embedding ----
    embed_k<<<R, 256, 0, stream>>>(idx, tok, pos, x, T, C);

    for (int l = 0; l < L; l++) {
        // LN1
        layernorm_k<<<R, 256, 0, stream>>>(x, ln1_g + (size_t)l * C,
                                           ln1_b + (size_t)l * C, xn, C);
        // QKV: per-head GEMM xn[R,C] @ W[h][C,HS] -> {q,k,v}[h][R][HS]
        const float* Wqkv[3] = {Wq, Wk, Wv};
        float*       qkv[3]  = {q, k, v};
        for (int i = 0; i < 3; i++) {
            gemm_wmma<<<dim3(1, R / BM, H), 128, 0, stream>>>(
                xn, Wqkv[i] + (size_t)l * H * C * HS, qkv[i], nullptr,
                R, HS, C, C, HS, HS,
                0LL, (long long)C * HS, (long long)R * HS, 0);
        }
        // attention, one batch element at a time (scores scratch = H*T*T)
        for (int b = 0; b < Bb; b++) {
            // S = q @ k^T   (B stored [T x HS] -> TRANSB)
            gemm_wmma<<<dim3(T / BN, T / BM, H), 128, 0, stream>>>(
                q + (size_t)b * T * HS, k + (size_t)b * T * HS, scores, nullptr,
                T, T, HS, HS, HS, T,
                (long long)R * HS, (long long)R * HS, (long long)T * T, GF_TRANSB);
            // causal softmax in place
            softmax_causal_k<<<H * T, 256, 0, stream>>>(scores, T, scale);
            // o_cat[b*T+t][h*64+s] = P @ v
            gemm_wmma<<<dim3(1, T / BM, H), 128, 0, stream>>>(
                scores, v + (size_t)b * T * HS,
                o_cat + (size_t)b * T * C, nullptr,
                T, HS, T, T, HS, C,
                (long long)T * T, (long long)R * HS, (long long)HS, 0);
        }
        // sum_h(o_h @ Ph_w[h] + Ph_b[h]) == o_cat @ vstack(Ph_w) + sum_h Ph_b
        headbias_sum_k<<<(C + 255) / 256, 256, 0, stream>>>(
            Ph_b + (size_t)l * H * C, bsum, H, C);
        gemm_wmma<<<dim3(C / BN, R / BM, 1), 128, 0, stream>>>(
            o_cat, Ph_w + (size_t)l * H * HS * C, sa_pre, bsum,
            R, C, C, C, C, C, 0LL, 0LL, 0LL, GF_BIAS);
        // x += sa_pre @ Pm_w + Pm_b
        gemm_wmma<<<dim3(C / BN, R / BM, 1), 128, 0, stream>>>(
            sa_pre, Pm_w + (size_t)l * C * C, x, Pm_b + (size_t)l * C,
            R, C, C, C, C, C, 0LL, 0LL, 0LL, GF_BIAS | GF_ADD);
        // LN2
        layernorm_k<<<R, 256, 0, stream>>>(x, ln2_g + (size_t)l * C,
                                           ln2_b + (size_t)l * C, xn, C);
        // MLP: h1 = gelu(xn @ W1 + b1)
        gemm_wmma<<<dim3(4 * C / BN, R / BM, 1), 128, 0, stream>>>(
            xn, W1 + (size_t)l * C * 4 * C, h1, b1 + (size_t)l * 4 * C,
            R, 4 * C, C, C, 4 * C, 4 * C, 0LL, 0LL, 0LL, GF_BIAS | GF_GELU);
        // x += h1 @ W2 + b2
        gemm_wmma<<<dim3(C / BN, R / BM, 1), 128, 0, stream>>>(
            h1, W2 + (size_t)l * 4 * C * C, x, b2 + (size_t)l * C,
            R, C, 4 * C, 4 * C, C, C, 0LL, 0LL, 0LL, GF_BIAS | GF_ADD);
    }

    // final LN + LM head
    layernorm_k<<<R, 256, 0, stream>>>(x, lnf_g, lnf_b, xn, C);
    gemm_wmma<<<dim3(V / BN, R / BM, 1), 128, 0, stream>>>(
        xn, head_w, logits, head_b,
        R, V, C, C, V, V, 0LL, 0LL, 0LL, GF_BIAS);
}
